// MixtralAttention_85341000171520
// MI455X (gfx1250) — matrix operand order repaired and checked by reading, not compiled, not run on hardware
//
#include <hip/hip_runtime.h>

typedef __attribute__((ext_vector_type(2)))  __bf16 v2bf;
typedef __attribute__((ext_vector_type(8)))  __bf16 v8bf;
typedef __attribute__((ext_vector_type(16))) __bf16 v16bf;
typedef __attribute__((ext_vector_type(4)))  __bf16 v4bf;
typedef __attribute__((ext_vector_type(8)))  float  v8f;
typedef __attribute__((ext_vector_type(4)))  unsigned int v4u;
typedef __attribute__((ext_vector_type(8)))  int v8i;
typedef __attribute__((ext_vector_type(4)))  int v4i;

#define T_TOK 2048
#define HID   4096
#define NH    32
#define NKV   8
#define HD    128
#define WIN   1024
#define QKV_N (HD * (NH + 2 * NKV))   // 6144
#define ATT_SCALE 0.08838834764831845f // 128^-0.5

static __device__ __forceinline__ v16bf cat16(v8bf lo, v8bf hi) {
  return __builtin_shufflevector(lo, hi, 0,1,2,3,4,5,6,7,8,9,10,11,12,13,14,15);
}

// ---- Tensor Data Mover: 2D bf16 tile (rows x cols), cols contiguous in memory,
// rows separated by stride_elems. Destination LDS is contiguous tile.
// D# layout per CDNA5 ISA; 6-arg builtin on this toolchain.
static __device__ __forceinline__ void tdm_load_2d_bf16(
    unsigned lds_off, const void* gptr, unsigned tile_cols, unsigned tile_rows,
    unsigned stride_elems, unsigned tensor_d0, unsigned tensor_d1) {
  unsigned long long ga = (unsigned long long)(size_t)gptr;
  v4u g0;
  g0.x = 1u;                                            // count=1, user mode
  g0.y = lds_off;                                       // lds_addr (bytes)
  g0.z = (unsigned)(ga & 0xffffffffu);                  // global_addr[31:0]
  g0.w = (unsigned)((ga >> 32) & 0x01ffffffu) | (2u << 30); // addr[56:32], type=2
  v8i g1;
  g1[0] = (int)(1u << 16);                              // data_size=1 (2 bytes)
  g1[1] = (int)((tensor_d0 & 0xffffu) << 16);           // tensor_dim0[15:0] @bit48
  g1[2] = (int)((tensor_d0 >> 16) | ((tensor_d1 & 0xffffu) << 16));
  g1[3] = (int)((tensor_d1 >> 16) | ((tile_cols & 0xffffu) << 16)); // tile_dim0
  g1[4] = (int)(tile_rows & 0xffffu);                   // tile_dim1 (tile_dim2=0)
  g1[5] = (int)stride_elems;                            // tensor_dim0_stride[31:0]
  g1[6] = 0;
  g1[7] = 0;
  v4i z4 = {0, 0, 0, 0};
  v8i z8 = {0, 0, 0, 0, 0, 0, 0, 0};
  __builtin_amdgcn_tensor_load_to_lds(g0, g1, z4, z4, z8, 0);
}

// ---- one-time fp32 -> bf16 weight conversion (4 elems/thread) ----
__global__ __launch_bounds__(256)
void cvt_f32_bf16(const float* __restrict__ in, __bf16* __restrict__ out) {
  int i = (blockIdx.x * 256 + threadIdx.x) * 4;
  float4 v = *(const float4*)(in + i);
  v4bf o;
  o[0] = (__bf16)v.x; o[1] = (__bf16)v.y; o[2] = (__bf16)v.z; o[3] = (__bf16)v.w;
  *(v4bf*)(out + i) = o;
}

// C[M,N] = A[M,K] (f32 row-major) x W[N,K]^T (bf16 row-major), C f32 row-major.
// Block: 256 threads = 8 waves. Block tile 128x128, wave tile 64x32, K-step 64.
// Double-buffered LDS: W tile k+1 staged by the TDM and A tile k+1 by VALU
// conversion while tile k is consumed by WMMA (16 WMMA per barrier pair).
template <int N, int K>
__global__ __launch_bounds__(256)
void gemm_wmma(const float* __restrict__ A, const __bf16* __restrict__ W,
               float* __restrict__ C) {
  __shared__ __bf16 As[2][128][64];
  __shared__ __bf16 Ws[2][128][64];

  const int tid  = threadIdx.x;
  const int lane = tid & 31;
  const int wave = tid >> 5;
  const int wm   = wave >> 2;   // 0..1  (64-row strip)
  const int wn   = wave & 3;    // 0..3  (32-col strip)
  const int bm   = blockIdx.y * 128;
  const int bn   = blockIdx.x * 128;

  const v8f vzero = {0.f,0.f,0.f,0.f,0.f,0.f,0.f,0.f};
  v8f acc[4][2];
#pragma unroll
  for (int i = 0; i < 4; i++)
#pragma unroll
    for (int j = 0; j < 2; j++) acc[i][j] = vzero;

  const int lrow = tid >> 1;           // 0..127
  const int lcol = (tid & 1) * 32;     // 0 or 32
  const float* ag = A + (size_t)(bm + lrow) * K + lcol;
  const __bf16* wtile0 = W + (size_t)bn * K;

  auto stage_a = [&](int buf, int k0) {
    __builtin_prefetch(ag + k0 + 128, 0, 1);
#pragma unroll
    for (int j = 0; j < 32; j += 4) {
      float4 va = *(const float4*)(ag + k0 + j);
      As[buf][lrow][lcol + j + 0] = (__bf16)va.x;
      As[buf][lrow][lcol + j + 1] = (__bf16)va.y;
      As[buf][lrow][lcol + j + 2] = (__bf16)va.z;
      As[buf][lrow][lcol + j + 3] = (__bf16)va.w;
    }
  };
  auto issue_w = [&](int buf, int k0) {
    tdm_load_2d_bf16((unsigned)(size_t)&Ws[buf][0][0], wtile0 + k0,
                     64, 128, K, K, N);
  };

  // prologue: fill buffer 0 with tile 0
  if (wave == 0) issue_w(0, 0);
  stage_a(0, 0);
  __builtin_amdgcn_s_wait_tensorcnt(0);
  __syncthreads();

  const int NT = K / 64;
  for (int kt = 0; kt < NT; kt++) {
    const int cur = kt & 1, nxt = cur ^ 1;
    if (kt + 1 < NT) {                 // stage tile k+1 while computing tile k
      if (wave == 0) issue_w(nxt, (kt + 1) * 64);
      stage_a(nxt, (kt + 1) * 64);
    }

#pragma unroll
    for (int c = 0; c < 2; c++) {      // two 32-wide K chunks
      const int cb = c * 32;
      // A fragments: lane group selects K {kb..+7, kb+16..+23}, kb=(lane>>4)*8
      v16bf afrag[4];
      const int arow_kb = cb + (lane >> 4) * 8;
#pragma unroll
      for (int i = 0; i < 4; i++) {
        int row = wm * 64 + i * 16 + (lane & 15);
        v8bf lo = *(const v8bf*)&As[cur][row][arow_kb];
        v8bf hi = *(const v8bf*)&As[cur][row][arow_kb + 16];
        afrag[i] = cat16(lo, hi);
      }
      // B fragments: lane holds col n = lane&15, K {kb..+7, kb+16..+23}, kb=(lane>>4)*8
      v16bf bfrag[2];
      const int bk = cb + (lane >> 4) * 8;
#pragma unroll
      for (int j = 0; j < 2; j++) {
        int col = wn * 32 + j * 16 + (lane & 15);
        bfrag[j] = cat16(*(const v8bf*)&Ws[cur][col][bk], *(const v8bf*)&Ws[cur][col][bk + 16]);
      }
#pragma unroll
      for (int i = 0; i < 4; i++)
#pragma unroll
        for (int j = 0; j < 2; j++)
          acc[i][j] = __builtin_amdgcn_wmma_f32_16x16x32_bf16(
              false, afrag[i], false, bfrag[j], (short)0, acc[i][j], false, false);
    }

    __builtin_amdgcn_s_wait_tensorcnt(0);
    __syncthreads();
  }

  const int half = lane >> 4;
  const int nn   = lane & 15;
#pragma unroll
  for (int i = 0; i < 4; i++)
#pragma unroll
    for (int j = 0; j < 2; j++) {
      float* cp = C + (size_t)(bm + wm * 64 + i * 16 + half * 8) * N
                    + (bn + wn * 32 + j * 16 + nn);
#pragma unroll
      for (int r = 0; r < 8; r++) cp[(size_t)r * N] = acc[i][j][r];
    }
}

// RoPE on q,k + bf16 conversion; v straight-converted. One thread per (t, head, d<64).
__global__ __launch_bounds__(256)
void rope_split_kernel(const float* __restrict__ qkv,
                       const float* __restrict__ cs, const float* __restrict__ sn,
                       __bf16* __restrict__ q, __bf16* __restrict__ k,
                       __bf16* __restrict__ v) {
  int idx = blockIdx.x * 256 + threadIdx.x;   // T * 48 * 64
  int d = idx & 63;
  int h = (idx >> 6) % (NH + 2 * NKV);
  int t = idx / (64 * (NH + 2 * NKV));
  if (t >= T_TOK) return;
  const float* row = qkv + (size_t)t * QKV_N;
  float c = cs[t * 64 + d], s = sn[t * 64 + d];
  if (h < NH) {
    const float* r2 = row + h * HD;
    float x1 = r2[d], x2 = r2[d + 64];
    __bf16* o = q + ((size_t)t * NH + h) * HD;
    o[d]      = (__bf16)(x1 * c - x2 * s);
    o[d + 64] = (__bf16)(x2 * c + x1 * s);
  } else if (h < NH + NKV) {
    int hh = h - NH;
    const float* r2 = row + NH * HD + hh * HD;
    float x1 = r2[d], x2 = r2[d + 64];
    __bf16* o = k + ((size_t)t * NKV + hh) * HD;
    o[d]      = (__bf16)(x1 * c - x2 * s);
    o[d + 64] = (__bf16)(x2 * c + x1 * s);
  } else {
    int hh = h - NH - NKV;
    const float* r2 = row + (NH + NKV) * HD + hh * HD;
    __bf16* o = v + ((size_t)t * NKV + hh) * HD;
    o[d]      = (__bf16)r2[d];
    o[d + 64] = (__bf16)r2[d + 64];
  }
}

// Flash attention with sliding window, GQA (4 q heads per kv head).
// Grid: (T/128, NH). Block 256 = 8 waves; each wave owns 16 q rows.
// Double-buffered: K tile (32x128 bf16) via TDM and transposed V tile via VALU
// for step kt+1 are staged while step kt is computed.
__global__ __launch_bounds__(256)
void attn_wmma_kernel(const __bf16* __restrict__ Q, const __bf16* __restrict__ Kc,
                      const __bf16* __restrict__ Vc, float* __restrict__ Oout) {
  __shared__ __bf16 Ks[2][32][HD];    // keys x d
  __shared__ __bf16 Vs[2][HD][32];    // d x keys (transposed)
  __shared__ __bf16 Ps[8][16][32];    // per-wave P staging

  const int tid  = threadIdx.x;
  const int lane = tid & 31;
  const int wave = tid >> 5;
  const int h    = blockIdx.y;
  const int kvh  = h >> 2;            // GROUPS = 4
  const int qb   = blockIdx.x * 128;
  const int qrow0 = qb + wave * 16;
  const int half = lane >> 4;
  const int nn   = lane & 15;

  // Q fragments for this wave's 16 rows, D=128 -> 4 K-chunks of 32.
  v16bf qf[4];
  {
    const __bf16* qp = Q + ((size_t)(qrow0 + nn) * NH + h) * HD;
    const int kb = half * 8;
#pragma unroll
    for (int f = 0; f < 4; f++) {
      v8bf lo = *(const v8bf*)(qp + f * 32 + kb);
      v8bf hi = *(const v8bf*)(qp + f * 32 + 16 + kb);
      qf[f] = cat16(lo, hi);
    }
  }

  const v8f vzero = {0.f,0.f,0.f,0.f,0.f,0.f,0.f,0.f};
  float mrow[8], lrow[8];
  v8f oacc[8];
#pragma unroll
  for (int r = 0; r < 8; r++) { mrow[r] = -1e30f; lrow[r] = 0.f; }
#pragma unroll
  for (int nf = 0; nf < 8; nf++) oacc[nf] = vzero;

  auto issue_k = [&](int buf, int kbase) {
    tdm_load_2d_bf16((unsigned)(size_t)&Ks[buf][0][0],
                     Kc + ((size_t)kbase * NKV + kvh) * HD,
                     HD, 32, NKV * HD, HD, (unsigned)(T_TOK * NKV));
  };
  // V transpose staging: thread owns key-pair (tid&15) x 8-d chunk (tid>>4),
  // packs 2 keys per 32-bit LDS store.
  auto stage_v = [&](int buf, int kbase) {
    int kp = (tid & 15) * 2;
    int d0 = (tid >> 4) * 8;
    const __bf16* vp0 = Vc + ((size_t)(kbase + kp) * NKV + kvh) * HD + d0;
    v8bf a = *(const v8bf*)vp0;
    v8bf b = *(const v8bf*)(vp0 + (size_t)NKV * HD);
#pragma unroll
    for (int j = 0; j < 8; j++) {
      v2bf t; t[0] = a[j]; t[1] = b[j];
      *(v2bf*)&Vs[buf][d0 + j][kp] = t;
    }
  };

  const int kt0   = ((qb >= WIN) ? (qb - WIN) : 0) >> 5;
  const int ktEnd = (qb + 128) >> 5;

  // prologue: fill buffer 0 with tile kt0
  if (wave == 0) issue_k(0, kt0 * 32);
  stage_v(0, kt0 * 32);
  __builtin_amdgcn_s_wait_tensorcnt(0);
  __syncthreads();

  for (int kt = kt0; kt < ktEnd; ++kt) {
    const int kbase = kt * 32;
    const int cur = (kt - kt0) & 1, nxt = cur ^ 1;
    if (kt + 1 < ktEnd) {              // stage next tile while computing
      if (wave == 0) issue_k(nxt, kbase + 32);
      stage_v(nxt, kbase + 32);
    }

    // S = Q * K^T : two 16x16 tiles (keys 0-15, 16-31), K-dim = D in 4 chunks.
    // Preload all 4 B-fragments per key tile, then run the WMMA chain so the
    // DS loads overlap WMMA issue instead of forcing dscnt==0 per WMMA.
    v8f s[2];
    const int bk = half * 8;
#pragma unroll
    for (int jf = 0; jf < 2; jf++) {
      int keycol = jf * 16 + nn;
      v16bf kf[4];
#pragma unroll
      for (int db = 0; db < 4; db++)
        kf[db] = cat16(*(const v8bf*)&Ks[cur][keycol][db * 32 + bk], *(const v8bf*)&Ks[cur][keycol][db * 32 + bk + 16]);
      v8f a = vzero;
#pragma unroll
      for (int db = 0; db < 4; db++)
        a = __builtin_amdgcn_wmma_f32_16x16x32_bf16(false, qf[db], false, kf[db],
                                                    (short)0, a, false, false);
      s[jf] = a;
    }

    // scale + window mask + per-row max (16-lane group reduction)
    float e[2][8], rmax[8];
#pragma unroll
    for (int r = 0; r < 8; r++) {
      int qi = qrow0 + half * 8 + r;
      float best = -1e30f;
#pragma unroll
      for (int jf = 0; jf < 2; jf++) {
        int kj = kbase + jf * 16 + nn;
        float v = s[jf][r] * ATT_SCALE;
        bool ok = (kj <= qi) && ((qi - kj) <= WIN);
        v = ok ? v : -1e30f;
        e[jf][r] = v;
        best = fmaxf(best, v);
      }
#pragma unroll
      for (int m = 1; m < 16; m <<= 1) best = fmaxf(best, __shfl_xor(best, m, 32));
      rmax[r] = best;
    }

    // online softmax update + stage P (bf16) to LDS
#pragma unroll
    for (int r = 0; r < 8; r++) {
      float mnew  = fmaxf(mrow[r], rmax[r]);
      float alpha = __expf(mrow[r] - mnew);
      mrow[r] = mnew;
      float rs = 0.f;
#pragma unroll
      for (int jf = 0; jf < 2; jf++) {
        float p = __expf(e[jf][r] - mnew);
        rs += p;
        Ps[wave][half * 8 + r][jf * 16 + nn] = (__bf16)p;
      }
#pragma unroll
      for (int m = 1; m < 16; m <<= 1) rs += __shfl_xor(rs, m, 32);
      lrow[r] = lrow[r] * alpha + rs;
#pragma unroll
      for (int nf = 0; nf < 8; nf++) oacc[nf][r] *= alpha;
    }

    // P fragment (A-layout, 16x32) from per-wave LDS; same-wave LDS is in-order.
    v16bf pf;
    {
      const int kb = half * 8;
      v8bf lo = *(const v8bf*)&Ps[wave][nn][kb];
      v8bf hi = *(const v8bf*)&Ps[wave][nn][kb + 16];
      pf = cat16(lo, hi);
    }
    // O += P * V : 8 d-column tiles, K-dim = 32 keys; fragments in groups of 4.
#pragma unroll
    for (int g = 0; g < 2; g++) {
      v16bf vf[4];
#pragma unroll
      for (int t = 0; t < 4; t++)
        vf[t] = cat16(*(const v8bf*)&Vs[cur][(g * 4 + t) * 16 + nn][half * 8], *(const v8bf*)&Vs[cur][(g * 4 + t) * 16 + nn][half * 8 + 16]);
#pragma unroll
      for (int t = 0; t < 4; t++)
        oacc[g * 4 + t] = __builtin_amdgcn_wmma_f32_16x16x32_bf16(
            false, pf, false, vf[t], (short)0, oacc[g * 4 + t], false, false);
    }

    __builtin_amdgcn_s_wait_tensorcnt(0);
    __syncthreads();
  }

  // normalize and store fp32 (T, NH*HD)
#pragma unroll
  for (int r = 0; r < 8; r++) {
    float inv = 1.0f / lrow[r];
    int row = qrow0 + half * 8 + r;
    float* orow = Oout + (size_t)row * (NH * HD) + h * HD;
#pragma unroll
    for (int nf = 0; nf < 8; nf++) orow[nf * 16 + nn] = oacc[nf][r] * inv;
  }
}

extern "C" void kernel_launch(void* const* d_in, const int* in_sizes, int n_in,
                              void* d_out, int out_size, void* d_ws, size_t ws_size,
                              hipStream_t stream) {
  const float* hs   = (const float*)d_in[0];  // (T, HID)
  const float* cs   = (const float*)d_in[1];  // (T, 64)
  const float* sn   = (const float*)d_in[2];  // (T, 64)
  const float* wqkv = (const float*)d_in[3];  // (6144, HID)
  const float* wo   = (const float*)d_in[4];  // (HID, HID)
  float* out = (float*)d_out;

  char* ws = (char*)d_ws;
  size_t off = 0;
  float*  qkv   = (float*)(ws + off);  off += (size_t)T_TOK * QKV_N * sizeof(float);
  __bf16* qb    = (__bf16*)(ws + off); off += (size_t)T_TOK * NH  * HD * sizeof(__bf16);
  __bf16* kb    = (__bf16*)(ws + off); off += (size_t)T_TOK * NKV * HD * sizeof(__bf16);
  __bf16* vb    = (__bf16*)(ws + off); off += (size_t)T_TOK * NKV * HD * sizeof(__bf16);
  float*  att   = (float*)(ws + off);  off += (size_t)T_TOK * NH  * HD * sizeof(float);
  __bf16* wqkvb = (__bf16*)(ws + off); off += (size_t)QKV_N * HID * sizeof(__bf16);
  __bf16* wob   = (__bf16*)(ws + off); off += (size_t)HID * HID * sizeof(__bf16);

  // 0) weights fp32 -> bf16
  cvt_f32_bf16<<<(QKV_N * HID) / (256 * 4), 256, 0, stream>>>(wqkv, wqkvb);
  cvt_f32_bf16<<<(HID * HID) / (256 * 4), 256, 0, stream>>>(wo, wob);

  // 1) QKV projection: (2048x4096) x (6144x4096)^T
  dim3 g1(QKV_N / 128, T_TOK / 128);
  gemm_wmma<QKV_N, HID><<<g1, 256, 0, stream>>>(hs, wqkvb, qkv);

  // 2) RoPE + split to bf16 q/k/v
  int rthreads = T_TOK * (NH + 2 * NKV) * 64;
  rope_split_kernel<<<rthreads / 256, 256, 0, stream>>>(qkv, cs, sn, qb, kb, vb);

  // 3) sliding-window GQA flash attention
  dim3 g3(T_TOK / 128, NH);
  attn_wmma_kernel<<<g3, 256, 0, stream>>>(qb, kb, vb, att);

  // 4) output projection: (2048x4096) x (4096x4096)^T
  dim3 g4(HID / 128, T_TOK / 128);
  gemm_wmma<HID, HID><<<g4, 256, 0, stream>>>(att, wob, out);
}
